// CausalMultiheadAttention_24404004176076
// MI455X (gfx1250) — matrix operand order repaired and checked by reading, not compiled, not run on hardware
//
#include <hip/hip_runtime.h>
#include <hip/hip_bf16.h>

// ---------------------------------------------------------------------------
// CDNA5 (gfx1250) causal MHA + RoPE, all GEMMs via v_wmma_f32_16x16x32_bf16.
// Round 2: software-pipelined GEMM (double-buffered operands, 32x64 wave
// tile) and 32-query attention waves with K-load pipelining.
// ---------------------------------------------------------------------------

typedef __attribute__((ext_vector_type(16))) __bf16 v16bf;
typedef __attribute__((ext_vector_type(8)))  float  v8f;

union V16 {
    v16bf v;
    uint4 u[2];
};

__device__ __forceinline__ unsigned short f32_to_bf16(float f) {
    unsigned int u = __float_as_uint(f);
    unsigned int r = u + 0x7FFFu + ((u >> 16) & 1u);   // round-to-nearest-even
    return (unsigned short)(r >> 16);
}
__device__ __forceinline__ float bf16_to_f32(unsigned short h) {
    return __uint_as_float(((unsigned int)h) << 16);
}

__device__ __forceinline__ v8f wmma_bf16(const V16& a, const V16& b, v8f c) {
    // 8 args: (neg_a, A, neg_b, B, c_mod, C, reuse_a, reuse_b)
    return __builtin_amdgcn_wmma_f32_16x16x32_bf16(
        false, a.v, false, b.v, (short)0, c, false, false);
}

// A-operand: lane holds 16 consecutive K values of its row (o = 16*hlf)
__device__ __forceinline__ void load_a(V16& a, const unsigned short* p) {
    a.u[0] = *(const uint4*)(p);
    a.u[1] = *(const uint4*)(p + 8);
}
// B-operand: lane holds 16 consecutive K values of its column
__device__ __forceinline__ void load_b(V16& b, const unsigned short* p) {
    b.u[0] = *(const uint4*)(p);
    b.u[1] = *(const uint4*)(p + 8);
}

// ---------------------------------------------------------------------------
// f32 -> bf16 elementwise (vec4)
// ---------------------------------------------------------------------------
__global__ void cvt_bf16_kernel(const float* __restrict__ in,
                                unsigned short* __restrict__ out, int n) {
    int i = (blockIdx.x * blockDim.x + threadIdx.x) * 4;
    if (i + 3 < n) {
        float4 f = *(const float4*)(in + i);
        uint2 p;
        p.x = (unsigned int)f32_to_bf16(f.x) | ((unsigned int)f32_to_bf16(f.y) << 16);
        p.y = (unsigned int)f32_to_bf16(f.z) | ((unsigned int)f32_to_bf16(f.w) << 16);
        *(uint2*)(out + i) = p;
    }
}

// ---------------------------------------------------------------------------
// f32 [rows][cols] -> bf16 transposed [cols][rows]  (LDS-tiled)
// ---------------------------------------------------------------------------
__global__ void transpose_cvt_kernel(const float* __restrict__ in,
                                     unsigned short* __restrict__ out,
                                     int rows, int cols) {
    __shared__ float tile[32][33];
    int bx = blockIdx.x * 32;   // over cols
    int by = blockIdx.y * 32;   // over rows
    int tx = threadIdx.x;       // 0..31
    int ty = threadIdx.y;       // 0..7
#pragma unroll
    for (int i = 0; i < 32; i += 8)
        tile[ty + i][tx] = in[(size_t)(by + ty + i) * cols + (bx + tx)];
    __syncthreads();
#pragma unroll
    for (int i = 0; i < 32; i += 8)
        out[(size_t)(bx + ty + i) * rows + (by + tx)] = f32_to_bf16(tile[tx][ty + i]);
}

// ---------------------------------------------------------------------------
// NT GEMM: C[M][N] = sum_k A[m][k] * BT[n][k], bf16 in, f32 accum.
// Block = 128 threads (4 waves); block tile 128(M) x 64(N);
// wave tile 32(M) x 64(N): 8 WMMA per k-step, double-buffered operands.
// ---------------------------------------------------------------------------
__global__ __launch_bounds__(128) void gemm_nt_kernel(
    const unsigned short* __restrict__ A, int lda,
    const unsigned short* __restrict__ BT, int ldb,
    unsigned short* __restrict__ Cb, float* __restrict__ Cf, int ldc,
    int K, int out_f32) {
    const int lane = threadIdx.x & 31;
    const int wave = threadIdx.x >> 5;
    const int l16 = lane & 15;
    const int hlf = (lane >> 4) & 1;

    const int mBase = blockIdx.y * 128 + wave * 32;
    const int nBase = blockIdx.x * 64;

    v8f acc[2][4] = {};

    const unsigned short* arow0 = A + (size_t)(mBase + l16) * lda + hlf * 16;
    const unsigned short* arow1 = arow0 + (size_t)16 * lda;
    const unsigned short* brow  = BT + (size_t)(nBase + l16) * ldb + hlf * 16;
    const size_t bcs = (size_t)16 * ldb;   // B column-chunk stride

    V16 aC[2], aN[2], bC[4], bN[4];

    // prime buffer "C" with k=0
    load_a(aC[0], arow0);
    load_a(aC[1], arow1);
#pragma unroll
    for (int c = 0; c < 4; ++c) load_b(bC[c], brow + c * bcs);

    for (int k0 = 0; k0 < K; k0 += 64) {
        // issue loads for k0+32 (K % 64 == 0 -> always in range)
        load_a(aN[0], arow0 + k0 + 32);
        load_a(aN[1], arow1 + k0 + 32);
#pragma unroll
        for (int c = 0; c < 4; ++c) load_b(bN[c], brow + c * bcs + k0 + 32);
        __builtin_prefetch(arow0 + k0 + 128, 0, 1);

#pragma unroll
        for (int c = 0; c < 4; ++c) {
            acc[0][c] = wmma_bf16(aC[0], bC[c], acc[0][c]);
            acc[1][c] = wmma_bf16(aC[1], bC[c], acc[1][c]);
        }

        if (k0 + 64 < K) {   // issue loads for k0+64 into buffer "C"
            load_a(aC[0], arow0 + k0 + 64);
            load_a(aC[1], arow1 + k0 + 64);
#pragma unroll
            for (int c = 0; c < 4; ++c) load_b(bC[c], brow + c * bcs + k0 + 64);
        }

#pragma unroll
        for (int c = 0; c < 4; ++c) {
            acc[0][c] = wmma_bf16(aN[0], bN[c], acc[0][c]);
            acc[1][c] = wmma_bf16(aN[1], bN[c], acc[1][c]);
        }
    }

    // C layout: VGPR r -> row (+ 8*hlf), col = l16
#pragma unroll
    for (int mt = 0; mt < 2; ++mt) {
#pragma unroll
        for (int c = 0; c < 4; ++c) {
            int col = nBase + c * 16 + l16;
#pragma unroll
            for (int r = 0; r < 8; ++r) {
                int row = mBase + mt * 16 + r + 8 * hlf;
                if (out_f32)
                    Cf[(size_t)row * ldc + col] = acc[mt][c][r];
                else
                    Cb[(size_t)row * ldc + col] = f32_to_bf16(acc[mt][c][r]);
            }
        }
    }
}

// ---------------------------------------------------------------------------
// RoPE in-place on bf16 Q and K: [B*S][1024], head dim 64, pairs (2i, 2i+1)
// ---------------------------------------------------------------------------
__global__ void rope_kernel(unsigned short* __restrict__ Q,
                            unsigned short* __restrict__ K,
                            const int* __restrict__ pos, int S) {
    int idx = blockIdx.x * blockDim.x + threadIdx.x;  // B*S*512 pairs
    int row = idx >> 9;        // token row (0..4095)
    int p   = idx & 511;
    int head = p >> 5;
    int i    = p & 31;
    int col  = head * 64 + 2 * i;
    int tpos = pos[row & (S - 1)];
    float inv = __powf(10000.0f, -(float)(2 * i) * (1.0f / 64.0f));
    float ang = (float)tpos * inv;
    float sn, cs;
    __sincosf(ang, &sn, &cs);
    size_t base = (size_t)row * 1024 + col;

    float q1 = bf16_to_f32(Q[base]), q2 = bf16_to_f32(Q[base + 1]);
    Q[base]     = f32_to_bf16(q1 * cs - q2 * sn);
    Q[base + 1] = f32_to_bf16(q1 * sn + q2 * cs);

    float k1 = bf16_to_f32(K[base]), k2 = bf16_to_f32(K[base + 1]);
    K[base]     = f32_to_bf16(k1 * cs - k2 * sn);
    K[base + 1] = f32_to_bf16(k1 * sn + k2 * cs);
}

// ---------------------------------------------------------------------------
// Flash attention: one wave per (batch, head, 32-query block).
// Two 16-row query tiles share every K/V operand load. K loads for the next
// j-tile are issued into the dead bK registers before softmax so VALU + LDS
// work hides their latency; V loads issue right after the score WMMAs.
// ---------------------------------------------------------------------------
__device__ __forceinline__ void load_bk4(V16* bK, const unsigned short* Kb,
                                         int rowbase, int colbase, int l16) {
    const unsigned short* k0 = Kb + (size_t)(rowbase + l16) * 1024 + colbase;
    const unsigned short* k1 = k0 + 16 * 1024;
    load_b(bK[0], k0);        // keys +0..15,  d 0..31
    load_b(bK[1], k0 + 32);   // keys +0..15,  d 32..63
    load_b(bK[2], k1);        // keys +16..31, d 0..31
    load_b(bK[3], k1 + 32);   // keys +16..31, d 32..63
}

__global__ __launch_bounds__(32) void attn_kernel(
    const unsigned short* __restrict__ Qb,   // [B*S][1024] bf16 (RoPE'd)
    const unsigned short* __restrict__ Kb,   // [B*S][1024] bf16 (RoPE'd)
    const unsigned short* __restrict__ Vt,   // [1024][B*S] bf16 (transposed V)
    unsigned short* __restrict__ Ob,         // [B*S][1024] bf16
    int S, int BS) {
    __shared__ unsigned short Pls[16][32];

    const int lane = threadIdx.x & 31;
    const int l16  = lane & 15;
    const int hlf  = (lane >> 4) & 1;

    const int qbase = blockIdx.x * 32;       // 32 queries per wave
    const int bh    = blockIdx.y;            // b*16 + h
    const int bIdx  = bh >> 4;
    const int h     = bh & 15;
    const int bs0   = bIdx * S;
    const int hbase = h * 64;

    // Q A-operands for both query tiles (k = d 0..31 and 32..63)
    V16 aQ[2][2];
#pragma unroll
    for (int t = 0; t < 2; ++t) {
        const unsigned short* qrow =
            Qb + (size_t)(bs0 + qbase + t * 16 + l16) * 1024 + hbase + hlf * 16;
        load_a(aQ[t][0], qrow);
        load_a(aQ[t][1], qrow + 32);
    }

    v8f acc[2][4] = {};
    float mrow[2][8], lrow[2][8];
#pragma unroll
    for (int t = 0; t < 2; ++t)
#pragma unroll
        for (int r = 0; r < 8; ++r) { mrow[t][r] = -3.0e38f; lrow[t][r] = 0.0f; }

    const float scale = 0.125f;  // 1/sqrt(64)
    const int jmax = qbase;      // last 32-key tile (keys <= qbase+31)

    V16 bK[4];
    load_bk4(bK, Kb, bs0, hbase + hlf * 16, l16);   // jbase = 0

    for (int jbase = 0; jbase <= jmax; jbase += 32) {
        // scores: S[t][js] over 32 keys, k = d = 64
        v8f Sv[2][2];
#pragma unroll
        for (int t = 0; t < 2; ++t)
#pragma unroll
            for (int js = 0; js < 2; ++js) {
                v8f z = {};
                z = wmma_bf16(aQ[t][0], bK[js * 2 + 0], z);
                z = wmma_bf16(aQ[t][1], bK[js * 2 + 1], z);
                Sv[t][js] = z;
            }

        // V B-operands for this tile (lane = d column, k = key)
        V16 bV[4];
#pragma unroll
        for (int c = 0; c < 4; ++c) {
            const unsigned short* vrow =
                Vt + (size_t)(hbase + c * 16 + l16) * (size_t)BS +
                bs0 + jbase + hlf * 16;
            load_b(bV[c], vrow);
        }

        // pipeline: next tile's K loads into the now-dead bK registers
        if (jbase + 32 <= jmax)
            load_bk4(bK, Kb, bs0 + jbase + 32, hbase + hlf * 16, l16);

#pragma unroll
        for (int t = 0; t < 2; ++t) {
            const int rbase = qbase + t * 16 + 8 * hlf;
            // scale + causal mask
#pragma unroll
            for (int r = 0; r < 8; ++r) {
                int rowg = rbase + r;
                float s0 = Sv[t][0][r] * scale;
                float s1 = Sv[t][1][r] * scale;
                if (jbase + l16 > rowg)      s0 = -1.0e30f;
                if (jbase + 16 + l16 > rowg) s1 = -1.0e30f;
                Sv[t][0][r] = s0; Sv[t][1][r] = s1;
            }
            // online softmax (row reductions within 16-lane half-groups)
#pragma unroll
            for (int r = 0; r < 8; ++r) {
                float v = fmaxf(Sv[t][0][r], Sv[t][1][r]);
                v = fmaxf(v, __shfl_xor(v, 1));
                v = fmaxf(v, __shfl_xor(v, 2));
                v = fmaxf(v, __shfl_xor(v, 4));
                v = fmaxf(v, __shfl_xor(v, 8));
                float mnew  = fmaxf(mrow[t][r], v);
                float alpha = __expf(mrow[t][r] - mnew);
                float p0 = __expf(Sv[t][0][r] - mnew);
                float p1 = __expf(Sv[t][1][r] - mnew);
                float rs = p0 + p1;
                rs += __shfl_xor(rs, 1);
                rs += __shfl_xor(rs, 2);
                rs += __shfl_xor(rs, 4);
                rs += __shfl_xor(rs, 8);
                lrow[t][r] = lrow[t][r] * alpha + rs;
                mrow[t][r] = mnew;
#pragma unroll
                for (int c = 0; c < 4; ++c) acc[t][c][r] *= alpha;
                Pls[8 * hlf + r][l16]      = f32_to_bf16(p0);
                Pls[8 * hlf + r][16 + l16] = f32_to_bf16(p1);
            }
            __syncthreads();

            // P as A-operand (16x32), re-laid-out through LDS
            V16 aP;
            load_a(aP, &Pls[l16][hlf * 16]);

#pragma unroll
            for (int c = 0; c < 4; ++c)
                acc[t][c] = wmma_bf16(aP, bV[c], acc[t][c]);
            __syncthreads();
        }
    }

    // epilogue: normalize, write O (C layout)
#pragma unroll
    for (int t = 0; t < 2; ++t)
#pragma unroll
        for (int c = 0; c < 4; ++c)
#pragma unroll
            for (int r = 0; r < 8; ++r) {
                float o = acc[t][c][r] / lrow[t][r];
                int rowt = bs0 + qbase + t * 16 + 8 * hlf + r;
                Ob[(size_t)rowt * 1024 + hbase + c * 16 + l16] = f32_to_bf16(o);
            }
}

// ---------------------------------------------------------------------------
extern "C" void kernel_launch(void* const* d_in, const int* in_sizes, int n_in,
                              void* d_out, int out_size, void* d_ws, size_t ws_size,
                              hipStream_t stream) {
    (void)in_sizes; (void)n_in; (void)out_size; (void)ws_size;

    const float* x  = (const float*)d_in[0];
    const float* Wq = (const float*)d_in[1];
    const float* Wk = (const float*)d_in[2];
    const float* Wv = (const float*)d_in[3];
    const float* Wo = (const float*)d_in[4];
    const int*  pos = (const int*)d_in[5];
    float* out = (float*)d_out;

    const int B = 2, S = 2048, D = 1024;
    const int M = B * S;  // 4096 tokens

    char* ws = (char*)d_ws;
    size_t off = 0;
    auto alloc = [&](size_t bytes) -> void* {
        void* p = ws + off;
        off += (bytes + 255) & ~(size_t)255;
        return p;
    };
    unsigned short* xb  = (unsigned short*)alloc((size_t)M * D * 2);
    unsigned short* WqT = (unsigned short*)alloc((size_t)D * D * 2);
    unsigned short* WkT = (unsigned short*)alloc((size_t)D * D * 2);
    unsigned short* WvT = (unsigned short*)alloc((size_t)D * D * 2);
    unsigned short* WoT = (unsigned short*)alloc((size_t)D * D * 2);
    unsigned short* Qb  = (unsigned short*)alloc((size_t)M * D * 2);
    unsigned short* Kbm = (unsigned short*)alloc((size_t)M * D * 2);
    unsigned short* Vtb = (unsigned short*)alloc((size_t)D * M * 2);
    unsigned short* Obm = (unsigned short*)alloc((size_t)M * D * 2);

    // 1) conversions
    cvt_bf16_kernel<<<(M * D) / (256 * 4), 256, 0, stream>>>(x, xb, M * D);
    dim3 tb(32, 8), tg(D / 32, D / 32);
    transpose_cvt_kernel<<<tg, tb, 0, stream>>>(Wq, WqT, D, D);
    transpose_cvt_kernel<<<tg, tb, 0, stream>>>(Wk, WkT, D, D);
    transpose_cvt_kernel<<<tg, tb, 0, stream>>>(Wv, WvT, D, D);
    transpose_cvt_kernel<<<tg, tb, 0, stream>>>(Wo, WoT, D, D);

    // 2) projections (all WMMA)
    // Q = x * Wq            : A = xb [4096x1024], BT = WqT
    gemm_nt_kernel<<<dim3(D / 64, M / 128), 128, 0, stream>>>(
        xb, D, WqT, D, Qb, nullptr, D, D, 0);
    // K = x * Wk
    gemm_nt_kernel<<<dim3(D / 64, M / 128), 128, 0, stream>>>(
        xb, D, WkT, D, Kbm, nullptr, D, D, 0);
    // V^T = Wv^T * x^T      : A = WvT [1024x1024], BT = xb [4096x1024], C = [1024x4096]
    gemm_nt_kernel<<<dim3(M / 64, D / 128), 128, 0, stream>>>(
        WvT, D, xb, D, Vtb, nullptr, M, D, 0);

    // 3) RoPE on Q, K
    rope_kernel<<<(M * 512) / 256, 256, 0, stream>>>(Qb, Kbm, pos, S);

    // 4) causal flash attention (32 queries per wave)
    attn_kernel<<<dim3(S / 32, B * 16), 32, 0, stream>>>(Qb, Kbm, Vtb, Obm, S, M);

    // 5) output projection, f32 epilogue straight to d_out
    gemm_nt_kernel<<<dim3(D / 64, M / 128), 128, 0, stream>>>(
        Obm, D, WoT, D, nullptr, out, D, D, 1);
}
